// EquivarianceNetwork_6923487281432
// MI455X (gfx1250) — matrix operand order repaired and checked by reading, not compiled, not run on hardware
//
#include <hip/hip_runtime.h>
#include <hip/hip_bf16.h>

typedef __bf16 bf16_t;
typedef __attribute__((ext_vector_type(16))) __bf16 v16bf;
typedef __attribute__((ext_vector_type(8)))  __bf16 v8bf;
typedef __attribute__((ext_vector_type(8)))  float  v8f;

union ABFrag { v16bf v; v8bf h[2]; };

// ---------------------------------------------------------------------------
// Transpose + f32->bf16 convert:  W[l][K][N] (f32, row-major) -> Wt[l][N][K]
// (bf16, row-major).  K-contiguous weight columns make the WMMA B-fragment a
// single contiguous 32B load per lane.
// ---------------------------------------------------------------------------
__global__ __launch_bounds__(256)
void transpose_convert_kernel(const float* __restrict__ W,
                              bf16_t* __restrict__ Wt,
                              int K, int N)
{
    __shared__ float tile[32][33];
    const int l  = blockIdx.z;
    const float* Win  = W  + (size_t)l * K * N;
    bf16_t*      Wout = Wt + (size_t)l * K * N;
    const int k0 = blockIdx.x * 32;
    const int n0 = blockIdx.y * 32;
    const int tx = threadIdx.x & 31;
    const int ty = threadIdx.x >> 5;   // 0..7

    #pragma unroll
    for (int i = 0; i < 4; ++i) {
        int k = k0 + ty + 8 * i;
        int n = n0 + tx;
        if (k < K && n < N) tile[ty + 8 * i][tx] = Win[(size_t)k * N + n];
    }
    __syncthreads();
    #pragma unroll
    for (int i = 0; i < 4; ++i) {
        int n = n0 + ty + 8 * i;
        int k = k0 + tx;
        if (n < N && k < K) Wout[(size_t)n * K + k] = (bf16_t)tile[tx][ty + 8 * i];
    }
}

// ---------------------------------------------------------------------------
// Gram kernel: scalars[b, n*16+m] = sum_d xr[b,n,d]*xr[b,m,d], output bf16.
// One wave per sample; the 4KB sample row is staged in LDS.
// ---------------------------------------------------------------------------
__global__ __launch_bounds__(256)
void gram_kernel(const float* __restrict__ x, bf16_t* __restrict__ scal)
{
    __shared__ float xs[8][1024];
    const int wave = threadIdx.x >> 5;
    const int lane = threadIdx.x & 31;
    const int b = blockIdx.x * 8 + wave;

    const float4* xrow = (const float4*)(x + (size_t)b * 1024);
    float4* dst = (float4*)xs[wave];
    #pragma unroll
    for (int i = lane; i < 256; i += 32) dst[i] = xrow[i];
    __syncthreads();

    #pragma unroll
    for (int e = lane; e < 256; e += 32) {
        const int n = e >> 4, m = e & 15;
        const float* xn = xs[wave] + n * 64;
        const float* xm = xs[wave] + m * 64;
        float s = 0.f;
        #pragma unroll
        for (int d = 0; d < 64; ++d) s += xn[d] * xm[d];
        scal[(size_t)b * 256 + e] = (bf16_t)s;
    }
}

// ---------------------------------------------------------------------------
// WMMA bf16 GEMM + bias + tanh:  C[M,N] = tanh(A[M,K] * Wt[N,K]^T + bias)
// Block: 256 threads = 8 waves, arranged 2(M) x 4(N).
// Block tile 128(M) x 256(N); wave tile 64 x 64 = 16 wmma accumulators.
// Per K-step: 4 A frags + 4 B frags -> 16 v_wmma (A frags shared across the
// 4 N-waves and B frags across the 2 M-waves via the WGP$).
// A-fragment (16x32): lane = row, half 0 -> K 0-15, half 1 -> K 16-31.
// B-fragment (32x16): lane = column, half 0 -> K 0-15, half 1 -> K 16-31
// (contiguous since Wt is [N,K]).
// ---------------------------------------------------------------------------
template <int KDIM>
__global__ __launch_bounds__(256)
void gemm_bias_tanh_kernel(const bf16_t* __restrict__ A,
                           const bf16_t* __restrict__ Wt,
                           const float*  __restrict__ bias,
                           bf16_t* __restrict__ C,
                           int N)
{
    const int wave = threadIdx.x >> 5;
    const int lane = threadIdx.x & 31;
    const int half = lane >> 4;
    const int lr   = lane & 15;

    const int m0 = blockIdx.x * 128 + (wave >> 2) * 64;   // wave M base
    const int n0 = blockIdx.y * 256 + (wave & 3) * 64;    // wave N base

    v8f acc[4][4];
    const v8f vzero = {0.f, 0.f, 0.f, 0.f, 0.f, 0.f, 0.f, 0.f};
    #pragma unroll
    for (int i = 0; i < 4; ++i)
        #pragma unroll
        for (int j = 0; j < 4; ++j) acc[i][j] = vzero;

    const bf16_t* arow  = A  + (size_t)(m0 + lr) * KDIM;
    const bf16_t* bcol0 = Wt + (size_t)(n0 + lr) * KDIM + half * 16;

    for (int k0 = 0; k0 < KDIM; k0 += 32) {
        ABFrag a[4];
        #pragma unroll
        for (int i = 0; i < 4; ++i) {
            const bf16_t* ap = arow + (size_t)(i * 16) * KDIM + k0;
            a[i].h[0] = *(const v8bf*)(ap +     half * 16);
            a[i].h[1] = *(const v8bf*)(ap + 8 + half * 16);
        }
        v16bf bfr[4];
        #pragma unroll
        for (int j = 0; j < 4; ++j)
            bfr[j] = *(const v16bf*)(bcol0 + (size_t)(j * 16) * KDIM + k0);

        #pragma unroll
        for (int i = 0; i < 4; ++i)
            #pragma unroll
            for (int j = 0; j < 4; ++j)
                acc[i][j] = __builtin_amdgcn_wmma_f32_16x16x32_bf16(
                    false, a[i].v, false, bfr[j], (short)0, acc[i][j],
                    false, false);
    }

    #pragma unroll
    for (int j = 0; j < 4; ++j) {
        const int n = n0 + j * 16 + lr;
        const float bv = bias[n];
        #pragma unroll
        for (int i = 0; i < 4; ++i) {
            #pragma unroll
            for (int r = 0; r < 8; ++r) {
                const int m = m0 + i * 16 + half * 8 + r;
                C[(size_t)m * N + n] = (bf16_t)tanhf(acc[i][j][r] + bv);
            }
        }
    }
}

// ---------------------------------------------------------------------------
// Head GEMM (N=16), f32 output, no tanh:  C[M,16] = A[M,K]*Wt[16,K]^T + bias
// 8 waves per block, each wave a 16x16 tile; block tile 128(M) x 16(N).
// ---------------------------------------------------------------------------
__global__ __launch_bounds__(256)
void gemm_bias_n16_kernel(const bf16_t* __restrict__ A,
                          const bf16_t* __restrict__ Wt,
                          const float*  __restrict__ bias,
                          float* __restrict__ C)
{
    constexpr int KDIM = 1024;
    const int wave = threadIdx.x >> 5;
    const int lane = threadIdx.x & 31;
    const int half = lane >> 4;
    const int lr   = lane & 15;

    const int m0 = blockIdx.x * 128 + wave * 16;

    v8f acc = {0.f, 0.f, 0.f, 0.f, 0.f, 0.f, 0.f, 0.f};
    const bf16_t* arow = A + (size_t)(m0 + lr) * KDIM;

    #pragma unroll 2
    for (int k0 = 0; k0 < KDIM; k0 += 32) {
        ABFrag a;
        a.h[0] = *(const v8bf*)(arow + k0 +     half * 16);
        a.h[1] = *(const v8bf*)(arow + k0 + 8 + half * 16);
        const bf16_t* bcol = Wt + (size_t)lr * KDIM + k0 + half * 16;
        v16bf bf = *(const v16bf*)bcol;
        acc = __builtin_amdgcn_wmma_f32_16x16x32_bf16(
            false, a.v, false, bf, (short)0, acc, false, false);
    }

    const float bv = bias[lr];
    #pragma unroll
    for (int r = 0; r < 8; ++r) {
        const int m = m0 + half * 8 + r;
        C[(size_t)m * 16 + lr] = acc[r] + bv;
    }
}

// ---------------------------------------------------------------------------
// Final recombine: out[b, l*64+d] = sum_n coeffs[l,b,n] * x[b, n*64+d]
// One block per sample; x row staged in LDS.
// ---------------------------------------------------------------------------
__global__ __launch_bounds__(256)
void combine_kernel(const float* __restrict__ coeffs,  // [16, B, 16]
                    const float* __restrict__ x,       // [B, 1024]
                    float* __restrict__ out,           // [B, 1024]
                    int B)
{
    __shared__ float xs[1024];
    const int b = blockIdx.x;
    for (int i = threadIdx.x; i < 1024; i += 256)
        xs[i] = x[(size_t)b * 1024 + i];
    __syncthreads();

    #pragma unroll
    for (int j = 0; j < 4; ++j) {
        const int idx = threadIdx.x + 256 * j;
        const int l = idx >> 6;
        const int d = idx & 63;
        const float* c = coeffs + ((size_t)l * B + b) * 16;
        float s = 0.f;
        #pragma unroll
        for (int n = 0; n < 16; ++n) s += c[n] * xs[n * 64 + d];
        out[(size_t)b * 1024 + idx] = s;
    }
}

// ---------------------------------------------------------------------------
extern "C" void kernel_launch(void* const* d_in, const int* in_sizes, int n_in,
                              void* d_out, int out_size, void* d_ws, size_t ws_size,
                              hipStream_t stream)
{
    const float* x  = (const float*)d_in[0];
    const float* W0 = (const float*)d_in[1];
    const float* b0 = (const float*)d_in[2];
    const float* W1 = (const float*)d_in[3];
    const float* b1 = (const float*)d_in[4];
    const float* W2 = (const float*)d_in[5];
    const float* b2 = (const float*)d_in[6];
    const float* W3 = (const float*)d_in[7];
    const float* b3 = (const float*)d_in[8];

    const int B = in_sizes[0] / 1024;   // 16384

    size_t off = 0;
    auto alloc = [&](size_t bytes) -> void* {
        void* p = (char*)d_ws + off;
        off += (bytes + 255) & ~(size_t)255;
        return p;
    };
    bf16_t* Wt0  = (bf16_t*)alloc(16ull * 1024 * 256  * 2);  // [l][1024][256]
    bf16_t* Wt1  = (bf16_t*)alloc(16ull * 1024 * 1024 * 2);  // [l][1024][1024]
    bf16_t* Wt2  = (bf16_t*)alloc(16ull * 1024 * 1024 * 2);
    bf16_t* Wt3  = (bf16_t*)alloc(16ull * 16   * 1024 * 2);  // [l][16][1024]
    bf16_t* scal = (bf16_t*)alloc((size_t)B * 256  * 2);     // [B,256]
    bf16_t* actA = (bf16_t*)alloc((size_t)B * 1024 * 2);     // ping
    bf16_t* actB = (bf16_t*)alloc((size_t)B * 1024 * 2);     // pong
    float*  coeffs = (float*)alloc(16ull * B * 16 * 4);      // [l][B][16]
    (void)ws_size; (void)n_in; (void)out_size;

    // Weight prep (transpose + bf16 convert)
    transpose_convert_kernel<<<dim3(8, 32, 16),  256, 0, stream>>>(W0, Wt0, 256,  1024);
    transpose_convert_kernel<<<dim3(32, 32, 16), 256, 0, stream>>>(W1, Wt1, 1024, 1024);
    transpose_convert_kernel<<<dim3(32, 32, 16), 256, 0, stream>>>(W2, Wt2, 1024, 1024);
    transpose_convert_kernel<<<dim3(32, 1, 16),  256, 0, stream>>>(W3, Wt3, 1024, 16);

    // Gram matrices -> bf16 scalars
    gram_kernel<<<B / 8, 256, 0, stream>>>(x, scal);

    // 16 independent MLPs, sequentially so per-group activations (32MB bf16)
    // stay resident in the 192MB L2.
    const dim3 gemm_grid(B / 128, 4);   // 128x256 block tiles over [16384, 1024]
    for (int l = 0; l < 16; ++l) {
        gemm_bias_tanh_kernel<256><<<gemm_grid, 256, 0, stream>>>(
            scal, Wt0 + (size_t)l * 1024 * 256, b0 + l * 1024, actA, 1024);
        gemm_bias_tanh_kernel<1024><<<gemm_grid, 256, 0, stream>>>(
            actA, Wt1 + (size_t)l * 1024 * 1024, b1 + l * 1024, actB, 1024);
        gemm_bias_tanh_kernel<1024><<<gemm_grid, 256, 0, stream>>>(
            actB, Wt2 + (size_t)l * 1024 * 1024, b2 + l * 1024, actA, 1024);
        gemm_bias_n16_kernel<<<dim3(B / 128), 256, 0, stream>>>(
            actA, Wt3 + (size_t)l * 16 * 1024, b3 + l * 16,
            coeffs + (size_t)l * B * 16);
    }

    // Recombine with xr
    combine_kernel<<<B, 256, 0, stream>>>(coeffs, x, (float*)d_out, B);
}
